// Tacotron2Decoder_33157147525605
// MI455X (gfx1250) — matrix operand order repaired and checked by reading, not compiled, not run on hardware
//
#include <hip/hip_runtime.h>
#include <hip/hip_bf16.h>

// ---------------- Problem constants ----------------
#define BB     32
#define LL     200
#define TT     500
#define VV     80
#define DM     256
#define DE     512
#define D4     1024
#define RR     2
#define STEPS  250
#define DA     128
#define KSIZE  31
#define PADC   15
#define NFILT  32
#define MAXSTEP 500
#define N4     4096
#define K1     1792   // pre2(256) | s1(1024) | g(512)
#define KT1    56     // K1/32
#define K2     2048   // s1n(1024) | s2(1024)
#define KT2    64
#define KPROJ  1536   // s2n(1024) | g(512)
#define NFR    160
#define ZONEOUT 0.1f

// ---------------- d_out layout (floats) ----------------
#define OUT_FRAMES 0                        // [32][500][80] = 1,280,000
#define OUT_TOKENS (32 * 500 * 80)          // [32][500]     = 16,000
#define OUT_ATTN   (OUT_TOKENS + 32 * 500)  // [32][500][200]= 3,200,000

// ---------------- d_ws layout (bytes) ----------------
// Packed-A buffers hold activations pre-swizzled into the WMMA A-fragment
// layout (bf16), written by the producer phases, streamed by the GEMM.
#define OFF_BAR   0u
#define OFF_A1P   256u                          // bf16 [2*KT1*512] = 114,688 B
#define OFF_A2P   (OFF_A1P + 2u*KT1*512u*2u)    // bf16 [2*KT2*512] = 131,072 B
#define OFF_PROJ  (OFF_A2P + 2u*KT2*512u*2u)    // f32 [32][1536] (s2n | g)
#define OFF_S1    (OFF_PROJ + 32u*KPROJ*4u)     // f32 [32][1024] (zoneout old s1)
#define OFF_C1    (OFF_S1  + 32u*D4*4u)         // f32 [32][1024]
#define OFF_C2    (OFF_C1  + 32u*D4*4u)         // f32 [32][1024]
#define OFF_CUM   (OFF_C2  + 32u*D4*4u)         // f32 [32][200]
#define ZERO_BYTES (OFF_CUM + 32u*LL*4u)
#define OFF_REC   ZERO_BYTES                    // f32 [32][4096]
#define OFF_ENC   (OFF_REC + 32u*N4*4u)         // f32 encproj [32][200][128]
#define OFF_W1P   (OFF_ENC + 32u*LL*DA*4u)      // bf16 [1792*4096]
#define OFF_W2P   (OFF_W1P + (unsigned)(K1)*N4*2u) // bf16 [2048*4096]

typedef __attribute__((ext_vector_type(16))) __bf16 v16bf;
typedef __attribute__((ext_vector_type(8)))  float  v8f;

// ---------------- helpers ----------------
__device__ __host__ __forceinline__ unsigned short f2bf_rne(float f) {
  unsigned u = __builtin_bit_cast(unsigned, f);
  unsigned r = u + 0x7FFFu + ((u >> 16) & 1u);
  return (unsigned short)(r >> 16);
}

__device__ __forceinline__ unsigned rng_hash(unsigned a, unsigned b, unsigned c) {
  unsigned x = a * 0x9E3779B9u ^ b * 0x85EBCA6Bu ^ c * 0xC2B2AE35u;
  x ^= x >> 16; x *= 0x7FEB352Du;
  x ^= x >> 15; x *= 0x846CA68Bu;
  x ^= x >> 16;
  return x;
}
__device__ __forceinline__ float rng_uniform(unsigned x) {
  return (float)(x >> 8) * (1.0f / 16777216.0f);
}
__device__ __forceinline__ float gate_fn(float x) {
  return tanhf(x * 0.5f) * 0.5f + 0.5f;   // == sigmoid(x), reference form
}

// Scatter one activation element into the bf16 WMMA A-fragment layout.
// Forward map (lane,h)->koff: koff = ((h&8)?h+8:h) + ((lane>=16)?8:0)
__device__ __forceinline__ void store_packA(unsigned short* apack, int KT,
                                            int m, int k, float v) {
  int mtile  = m >> 4;
  int kt     = k >> 5;
  int koff   = k & 31;
  int lanehi = (koff >> 3) & 1;
  int h      = (koff & 7) + ((koff & 16) ? 8 : 0);
  int lane   = (m & 15) + (lanehi << 4);
  apack[(((long)(mtile * KT + kt)) * 32 + lane) * 16 + h] = f2bf_rne(v);
}

// Sense-reversing grid-wide barrier (32 WGs, all resident).
__device__ __forceinline__ void grid_sync(unsigned* bar) {
  __syncthreads();
  __threadfence();
  if (threadIdx.x == 0) {
    volatile unsigned* gen = bar + 1;
    unsigned g = *gen;
    if (atomicAdd(bar, 1u) == gridDim.x - 1u) {
      bar[0] = 0u;
      __threadfence();
      atomicAdd((unsigned*)gen, 1u);
    } else {
      while (*gen == g) { __builtin_amdgcn_s_sleep(1); }
    }
  }
  __syncthreads();
  __threadfence();
}

// ---------------- prologue: pack weights to bf16 WMMA B-fragment layout ----
// frag layout: idx = ((ntile*KT + kt)*32 + lane)*16 + h
//   n = ntile*16 + (lane&15) ; k = kt*32 + (h&7) + ((lane&16)?8:0) + ((h&8)?16:0)
__global__ void pack_weights_kernel(const float* __restrict__ Wys,
                                    const float* __restrict__ Wss,
                                    const float* __restrict__ Wgs,
                                    const float* __restrict__ W2is,
                                    const float* __restrict__ W2ss,
                                    unsigned short* __restrict__ w1p,
                                    unsigned short* __restrict__ w2p) {
  const long tot1 = (long)K1 * N4;
  const long tot2 = (long)K2 * N4;
  long stride = (long)gridDim.x * blockDim.x;
  for (long i = (long)blockIdx.x * blockDim.x + threadIdx.x; i < tot1 + tot2; i += stride) {
    if (i < tot1) {
      long t = i; int h = (int)(t & 15); t >>= 4;
      int lane = (int)(t & 31); t >>= 5;
      int kt = (int)(t % KT1); int ntile = (int)(t / KT1);
      int k = kt * 32 + (h & 7) + ((lane & 16) ? 8 : 0) + ((h & 8) ? 16 : 0);
      int n = ntile * 16 + (lane & 15);
      float v;
      if (k < 256)        v = Wys[(long)k * N4 + n];
      else if (k < 1280)  v = Wss[(long)(k - 256) * N4 + n];
      else                v = Wgs[(long)(k - 1280) * N4 + n];
      w1p[i] = f2bf_rne(v);
    } else {
      long j = i - tot1;
      long t = j; int h = (int)(t & 15); t >>= 4;
      int lane = (int)(t & 31); t >>= 5;
      int kt = (int)(t % KT2); int ntile = (int)(t / KT2);
      int k = kt * 32 + (h & 7) + ((lane & 16) ? 8 : 0) + ((h & 8) ? 16 : 0);
      int n = ntile * 16 + (lane & 15);
      float v = (k < 1024) ? W2is[(long)k * N4 + n] : W2ss[(long)(k - 1024) * N4 + n];
      w2p[j] = f2bf_rne(v);
    }
  }
}

// ---------------- prologue: encproj = e_outputs @ W_encproj + b ------------
__global__ void encproj_kernel(const float* __restrict__ e_out,
                               const float* __restrict__ W, const float* __restrict__ bias,
                               float* __restrict__ out) {
  int idx = blockIdx.x * blockDim.x + threadIdx.x;          // 32*200*128
  if (idx >= BB * LL * DA) return;
  int da = idx & (DA - 1);
  int bl = idx >> 7;
  const float* e = e_out + (long)bl * DE;
  float acc = bias[da];
  for (int k = 0; k < DE; ++k) acc += e[k] * W[(long)k * DA + da];
  out[idx] = acc;
}

// ---------------- main persistent decoder ----------------
struct Params {
  const float *mel, *e_out, *conv_w, *W_convproj, *W_decproj, *W_self;
  const float *W_pre1, *b_pre1, *W_pre2, *b_pre2, *b_l1, *b_l2;
  const float *W_frame, *b_frame, *W_token, *b_token;
  const unsigned short *w1p, *w2p;
  const float *encproj;
  unsigned short *a1pack, *a2pack;
  float *proj, *rec, *s1, *c1, *c2, *cum_alpha;
  float *out_frames, *out_tokens, *out_attn;
  unsigned *bar;
};

// rec = A(bf16 fragments) @ W(bf16 fragments) + bias ; 256 waves, 2 N-tiles each
__device__ __forceinline__ void gemm_wmma(const unsigned short* __restrict__ apack,
                                          const unsigned short* __restrict__ wpack,
                                          const float* __restrict__ bias,
                                          float* __restrict__ rec, int KT) {
  int wid  = (blockIdx.x * blockDim.x + threadIdx.x) >> 5;   // 0..255
  int lane = threadIdx.x & 31;
  int mtile = wid >> 7;            // 0..1 (rows 0-15 / 16-31)
  int ngrp  = wid & 127;           // 0..127 -> cols ngrp*32 .. +31
  int n0 = ngrp * 32;
  v8f c0 = {}; v8f c1 = {};
  const unsigned short* ap  = apack + ((long)mtile * KT * 32 + lane) * 16;
  const unsigned short* bp0 = wpack + ((((long)(n0 >> 4)) * KT) * 32 + lane) * 16;
  const unsigned short* bp1 = wpack + ((((long)(n0 >> 4) + 1) * KT) * 32 + lane) * 16;
#pragma unroll 2
  for (int kt = 0; kt < KT; ++kt) {
    v16bf Av  = *(const v16bf*)(ap  + (long)kt * 512);
    v16bf Bv0 = *(const v16bf*)(bp0 + (long)kt * 512);
    v16bf Bv1 = *(const v16bf*)(bp1 + (long)kt * 512);
    __builtin_prefetch(bp0 + (long)(kt + 2) * 512, 0, 1);
    c0 = __builtin_amdgcn_wmma_f32_16x16x32_bf16(false, Av, false, Bv0,
                                                 (short)0, c0, false, false);
    c1 = __builtin_amdgcn_wmma_f32_16x16x32_bf16(false, Av, false, Bv1,
                                                 (short)0, c1, false, false);
  }
  int col = n0 + (lane & 15);
  int rbase = mtile * 16 + (lane >> 4) * 8;
#pragma unroll
  for (int r = 0; r < 8; ++r) {
    rec[(long)(rbase + r) * N4 + col]      = c0[r] + bias[col];
    rec[(long)(rbase + r) * N4 + col + 16] = c1[r] + bias[col + 16];
  }
}

// LSTM gate phase: writes new c (fp32), new s into fp32 state + packed-A slots.
__device__ void lstm_gates(const float* rec, float* cbuf,
                           float* sbuf, int sstride,
                           unsigned short* packA, int KTa, int kbaseA,
                           unsigned short* packB, int KTb, int kbaseB,
                           int step, unsigned salt) {
  int gsz = gridDim.x * blockDim.x;
  for (int idx = blockIdx.x * blockDim.x + threadIdx.x; idx < BB * D4; idx += gsz) {
    int b = idx >> 10, j = idx & 1023;
    const float* rb = rec + (long)b * N4;
    float ig = rb[j], fg = rb[1024 + j], cg = rb[2048 + j], og = rb[3072 + j];
    float c  = cbuf[idx];
    float so = sbuf[(long)b * sstride + j];
    float m  = (rng_uniform(rng_hash(step, (unsigned)idx, salt)) < ZONEOUT) ? 1.f : 0.f;
    float cn = m * c + (1.f - m) * (gate_fn(fg) * c + gate_fn(ig) * tanhf(cg));
    float sn = m * so + (1.f - m) * (gate_fn(og) * tanhf(cn));
    cbuf[idx] = cn;
    sbuf[(long)b * sstride + j] = sn;
    store_packA(packA, KTa, b, kbaseA + j, sn);
    if (packB) store_packA(packB, KTb, b, kbaseB + j, sn);
  }
}

__device__ void attn_prenet(const Params& p, int step) {
  __shared__ float prev[VV], pre1s[DM], decp[DA], cpad[LL + 2 * PADC];
  __shared__ float sc[LL], ex[LL], red0, red1;
  int b = blockIdx.x;                 // gridDim.x == 32 == batch
  int t = threadIdx.x;

  for (int v = t; v < VV; v += 256)
    prev[v] = (step == 0) ? 0.f : p.mel[((long)b * TT + (step * 2 - 1)) * VV + v];
  for (int i = t; i < LL + 2 * PADC; i += 256) cpad[i] = 0.f;
  __syncthreads();
  for (int l = t; l < LL; l += 256) cpad[PADC + l] = p.cum_alpha[b * LL + l];
  __syncthreads();

  // prenet layer 1 (dropout 0.5)
  for (int j = t; j < DM; j += 256) {
    float a = p.b_pre1[j];
    for (int v = 0; v < VV; ++v) a += prev[v] * p.W_pre1[v * DM + j];
    a = fmaxf(a, 0.f);
    a = (rng_uniform(rng_hash(step * 4u + 0u, (unsigned)(b * DM + j), 0x1234567u)) < 0.5f)
          ? a * 2.f : 0.f;
    pre1s[j] = a;
  }
  // decproj = s2 @ W_decproj   (s2 fp32 lives at proj[:,0:1024])
  const float* s2 = p.proj + (long)b * KPROJ;
  for (int da = t; da < DA; da += 256) {
    float a = 0.f;
    for (int k = 0; k < D4; ++k) a += s2[k] * p.W_decproj[k * DA + da];
    decp[da] = a;
  }
  __syncthreads();
  // prenet layer 2 -> packed A1 (k = 0..255)
  for (int j = t; j < DM; j += 256) {
    float a = p.b_pre2[j];
    for (int v = 0; v < DM; ++v) a += pre1s[v] * p.W_pre2[v * DM + j];
    a = fmaxf(a, 0.f);
    a = (rng_uniform(rng_hash(step * 4u + 1u, (unsigned)(b * DM + j), 0x89abcdeu)) < 0.5f)
          ? a * 2.f : 0.f;
    store_packA(p.a1pack, KT1, b, j, a);
  }

  // location-sensitive attention scores
  for (int l = t; l < LL; l += 256) {
    float ac[NFILT];
#pragma unroll
    for (int f = 0; f < NFILT; ++f) {
      float s = 0.f;
      for (int kk = 0; kk < KSIZE; ++kk) s += p.conv_w[f * KSIZE + kk] * cpad[l + kk];
      ac[f] = s;
    }
    const float* enc = p.encproj + ((long)b * LL + l) * DA;
    float s = 0.f;
    for (int da = 0; da < DA; ++da) {
      float v = decp[da] + enc[da];
#pragma unroll
      for (int f = 0; f < NFILT; ++f) v += ac[f] * p.W_convproj[f * DA + da];
      s += tanhf(v) * p.W_self[da];
    }
    sc[l] = s;
  }
  __syncthreads();
  if (t == 0) { float m = -1e30f; for (int l = 0; l < LL; ++l) m = fmaxf(m, sc[l]); red0 = m; }
  __syncthreads();
  for (int l = t; l < LL; l += 256) ex[l] = expf(sc[l] - red0);
  __syncthreads();
  if (t == 0) { float s = 0.f; for (int l = 0; l < LL; ++l) s += ex[l]; red1 = s; }
  __syncthreads();
  for (int l = t; l < LL; l += 256) {
    float al = ex[l] / red1;
    sc[l] = al;
    p.cum_alpha[b * LL + l] = cpad[PADC + l] + al;
    p.out_attn[((long)b * MAXSTEP + step) * LL + l] = al;
  }
  __syncthreads();
  // context g -> packed A1 (k = 1280..1791) and proj[:,1024:1536] (fp32)
  for (int d = t; d < DE; d += 256) {
    float s = 0.f;
    for (int l = 0; l < LL; ++l) s += sc[l] * p.e_out[((long)b * LL + l) * DE + d];
    store_packA(p.a1pack, KT1, b, DM + D4 + d, s);
    p.proj[(long)b * KPROJ + D4 + d] = s;
  }
}

__device__ void frame_token(const Params& p, int step) {
  int gsz = gridDim.x * blockDim.x;
  for (int idx = blockIdx.x * blockDim.x + threadIdx.x; idx < BB * (NFR + RR); idx += gsz) {
    int b = idx / (NFR + RR);
    int n = idx % (NFR + RR);
    const float* pr = p.proj + (long)b * KPROJ;
    if (n < NFR) {
      float a = p.b_frame[n];
      for (int k = 0; k < KPROJ; ++k) a += pr[k] * p.W_frame[(long)k * NFR + n];
      int rr = n / VV, vv = n % VV;
      p.out_frames[((long)b * MAXSTEP + (step * RR + rr)) * VV + vv] = a;
    } else {
      int r = n - NFR;
      float a = p.b_token[r];
      for (int k = 0; k < KPROJ; ++k) a += pr[k] * p.W_token[k * RR + r];
      p.out_tokens[(long)b * MAXSTEP + step * RR + r] = a;
    }
  }
}

__global__ void __launch_bounds__(256) decoder_kernel(Params p) {
  for (int step = 0; step < STEPS; ++step) {
    attn_prenet(p, step);                                       // P1
    grid_sync(p.bar);
    gemm_wmma(p.a1pack, p.w1p, p.b_l1, p.rec, KT1);             // P2: LSTM1 GEMM
    grid_sync(p.bar);
    lstm_gates(p.rec, p.c1, p.s1, D4,                           // P3: s1n
               p.a1pack, KT1, DM,                               //   -> A1 s1 slot (next step)
               p.a2pack, KT2, 0, step, 0xA11CEu);               //   -> A2 k=0..1023
    grid_sync(p.bar);
    gemm_wmma(p.a2pack, p.w2p, p.b_l2, p.rec, KT2);             // P4: LSTM2 GEMM
    grid_sync(p.bar);
    lstm_gates(p.rec, p.c2, p.proj, KPROJ,                      // P5: s2n -> proj[:,0:1024]
               p.a2pack, KT2, D4,                               //   -> A2 k=1024..2047 (next)
               (unsigned short*)0, 0, 0, step, 0xB0B0Bu);
    grid_sync(p.bar);
    frame_token(p, step);                                       // P6
    grid_sync(p.bar);
  }
}

// ---------------- host ----------------
extern "C" void kernel_launch(void* const* d_in, const int* in_sizes, int n_in,
                              void* d_out, int out_size, void* d_ws, size_t ws_size,
                              hipStream_t stream) {
  (void)in_sizes; (void)n_in; (void)ws_size;
  char* ws = (char*)d_ws;
  float* out = (float*)d_out;

  // zero outputs + recurrent state (incl. packed-A s1/s2 slots) + barrier
  hipMemsetAsync(d_out, 0, (size_t)out_size * sizeof(float), stream);
  hipMemsetAsync(d_ws, 0, (size_t)ZERO_BYTES, stream);

  const float* mel       = (const float*)d_in[0];
  const float* e_out     = (const float*)d_in[1];
  const float* conv_w    = (const float*)d_in[2];
  const float* W_convprj = (const float*)d_in[3];
  const float* W_encproj = (const float*)d_in[4];
  const float* b_encproj = (const float*)d_in[5];
  const float* W_decproj = (const float*)d_in[6];
  const float* W_self    = (const float*)d_in[7];
  const float* W_pre1    = (const float*)d_in[8];
  const float* b_pre1    = (const float*)d_in[9];
  const float* W_pre2    = (const float*)d_in[10];
  const float* b_pre2    = (const float*)d_in[11];
  const float* W_l1_ys   = (const float*)d_in[12];
  const float* W_l1_ss   = (const float*)d_in[13];
  const float* W_l1_gs   = (const float*)d_in[14];
  const float* b_l1_gs   = (const float*)d_in[15];
  const float* W_l2_is   = (const float*)d_in[16];
  const float* W_l2_ss   = (const float*)d_in[17];
  const float* b_l2_ss   = (const float*)d_in[18];
  const float* W_frame   = (const float*)d_in[19];
  const float* b_frame   = (const float*)d_in[20];
  const float* W_token   = (const float*)d_in[21];
  const float* b_token   = (const float*)d_in[22];

  unsigned short* w1p = (unsigned short*)(ws + OFF_W1P);
  unsigned short* w2p = (unsigned short*)(ws + OFF_W2P);
  float* encproj = (float*)(ws + OFF_ENC);

  pack_weights_kernel<<<4096, 256, 0, stream>>>(W_l1_ys, W_l1_ss, W_l1_gs,
                                                W_l2_is, W_l2_ss, w1p, w2p);
  encproj_kernel<<<(BB * LL * DA + 255) / 256, 256, 0, stream>>>(e_out, W_encproj,
                                                                 b_encproj, encproj);

  Params p;
  p.mel = mel; p.e_out = e_out; p.conv_w = conv_w; p.W_convproj = W_convprj;
  p.W_decproj = W_decproj; p.W_self = W_self;
  p.W_pre1 = W_pre1; p.b_pre1 = b_pre1; p.W_pre2 = W_pre2; p.b_pre2 = b_pre2;
  p.b_l1 = b_l1_gs; p.b_l2 = b_l2_ss;
  p.W_frame = W_frame; p.b_frame = b_frame; p.W_token = W_token; p.b_token = b_token;
  p.w1p = w1p; p.w2p = w2p; p.encproj = encproj;
  p.a1pack = (unsigned short*)(ws + OFF_A1P);
  p.a2pack = (unsigned short*)(ws + OFF_A2P);
  p.proj = (float*)(ws + OFF_PROJ);
  p.s1   = (float*)(ws + OFF_S1);
  p.rec  = (float*)(ws + OFF_REC);
  p.c1   = (float*)(ws + OFF_C1);
  p.c2   = (float*)(ws + OFF_C2);
  p.cum_alpha = (float*)(ws + OFF_CUM);
  p.out_frames = out + OUT_FRAMES;
  p.out_tokens = out + OUT_TOKENS;
  p.out_attn   = out + OUT_ATTN;
  p.bar = (unsigned*)(ws + OFF_BAR);

  decoder_kernel<<<BB, 256, 0, stream>>>(p);
}
